// RIPE_11373073400432
// MI455X (gfx1250) — matrix operand order repaired and checked by reading, not compiled, not run on hardware
//
#include <hip/hip_runtime.h>
#include <hip/hip_bf16.h>
#include <math.h>

// ---------------------------------------------------------------------------
// detectAndCompute pipeline for MI455X (gfx1250, wave32, WMMA)
// ---------------------------------------------------------------------------

typedef __attribute__((ext_vector_type(16))) _Float16 v16h;
typedef __attribute__((ext_vector_type(8)))  _Float16 v8h;
typedef __attribute__((ext_vector_type(8)))  float    v8f;

#define HM_W 2048
#define HM_H 2048
#define FC   512      // feat channels (GEMM K)
#define FH   256
#define FW   256
#define DC   256      // descriptor dim (GEMM N)
#define TOPK 2048
#define MAX_CAND (1 << 20)
#define THRESH 0.5f

struct WS {
  int      cand_count;
  int      gmax_bits;
  int      skip;
  unsigned prefix;
  int      rem;
  unsigned sel_bits;
  int      needed_eq;
  int      cnt_hi;
  int      cnt_eq;
  int      pad[7];
  int      hist[4096];
  float    sel_score[TOPK];
  int      sel_idx[TOPK];
};

__device__ __forceinline__ int iclamp(int v, int lo, int hi) {
  return v < lo ? lo : (v > hi ? hi : v);
}

// ---------------------------------------------------------------------------
// 0) init workspace
// ---------------------------------------------------------------------------
__global__ void k_init(WS* ws) {
  int i = blockIdx.x * blockDim.x + threadIdx.x;
  if (i < 4096) ws->hist[i] = 0;
  if (i < TOPK) {
    ws->sel_score[i] = __int_as_float(0xFF800000);  // -inf
    ws->sel_idx[i]   = 0;
  }
  if (i == 0) {
    ws->cand_count = 0;
    ws->gmax_bits  = (int)0xFF800000;   // -inf bits (global max is positive)
    ws->skip       = 0;
    ws->prefix     = 0;
    ws->rem        = 0;
    ws->sel_bits   = 0;
    ws->needed_eq  = 0;
    ws->cnt_hi     = 0;
    ws->cnt_eq     = 0;
  }
}

// ---------------------------------------------------------------------------
// 1) 3x3 NMS + threshold -> candidate append; global max of heatmap
// ---------------------------------------------------------------------------
__global__ __launch_bounds__(256)
void k_nms(const float* __restrict__ h, WS* ws,
           float* __restrict__ cand_s, int* __restrict__ cand_i) {
  __shared__ int bmax;
  const int x = blockIdx.x * 32 + threadIdx.x;
  const int y = blockIdx.y * 8  + threadIdx.y;
  if (threadIdx.x == 0 && threadIdx.y == 0) bmax = (int)0xFF800000;
  __syncthreads();

  const float c = h[y * HM_W + x];
  atomicMax(&bmax, __float_as_int(c));   // all positives order correctly

  float m = c;
#pragma unroll
  for (int dy = -1; dy <= 1; ++dy) {
    const int yy = y + dy;
    if (yy < 0 || yy >= HM_H) continue;
#pragma unroll
    for (int dx = -1; dx <= 1; ++dx) {
      const int xx = x + dx;
      if (xx < 0 || xx >= HM_W) continue;
      m = fmaxf(m, h[yy * HM_W + xx]);
    }
  }
  if (c == m && c > THRESH) {
    int p = atomicAdd(&ws->cand_count, 1);
    if (p < MAX_CAND) {
      cand_s[p] = c;
      cand_i[p] = y * HM_W + x;
    }
  }
  __syncthreads();
  if (threadIdx.x == 0 && threadIdx.y == 0) atomicMax(&ws->gmax_bits, bmax);
}

// ---------------------------------------------------------------------------
// 2) 3-pass radix select of the exact TOPK-th score (positive-float bits
//    are order-preserving as unsigned ints)
// ---------------------------------------------------------------------------
__global__ void k_hist(WS* ws, const float* __restrict__ cs, int pass) {
  if (pass > 0 && ws->skip) return;
  int n = ws->cand_count; if (n > MAX_CAND) n = MAX_CAND;
  const unsigned pref = ws->prefix;
  for (int i = blockIdx.x * blockDim.x + threadIdx.x; i < n;
       i += gridDim.x * blockDim.x) {
    const unsigned bits = __float_as_uint(cs[i]);
    if (pass == 0) {
      atomicAdd(&ws->hist[bits >> 20], 1);
    } else if (pass == 1) {
      if ((bits >> 20) == pref) atomicAdd(&ws->hist[(bits >> 8) & 0xFFFu], 1);
    } else {
      if ((bits >> 8) == pref) atomicAdd(&ws->hist[bits & 0xFFu], 1);
    }
  }
}

__global__ void k_scan(WS* ws, int pass) {
  if (pass > 0 && ws->skip) return;
  const int K  = (pass == 0) ? TOPK : ws->rem;
  const int nb = (pass == 0) ? 2048 : (pass == 1 ? 4096 : 256);
  long acc = 0; int found = -1; int rem = 0;
  for (int b = nb - 1; b >= 0; --b) {
    const int hv = ws->hist[b];
    if (found < 0 && acc + (long)hv >= (long)K) { found = b; rem = K - (int)acc; }
    acc += hv;
  }
  if (found < 0) {                 // fewer than TOPK candidates: take them all
    ws->skip = 1;
    ws->sel_bits = 0x3F000000u;    // bits(0.5f); every candidate is strictly >
    ws->needed_eq = 0;
  } else if (pass == 0) {
    ws->prefix = (unsigned)found;  ws->rem = rem;
  } else if (pass == 1) {
    ws->prefix = (ws->prefix << 12) | (unsigned)found;  ws->rem = rem;
  } else {
    ws->sel_bits  = (ws->prefix << 8) | (unsigned)found;
    ws->needed_eq = rem;
  }
  for (int b = 0; b < 4096; ++b) ws->hist[b] = 0;
}

// ---------------------------------------------------------------------------
// 3) compact the exact TOPK set into sel_score/sel_idx
// ---------------------------------------------------------------------------
__global__ void k_compact(WS* ws, const float* __restrict__ cs,
                          const int* __restrict__ ci) {
  int n = ws->cand_count; if (n > MAX_CAND) n = MAX_CAND;
  const unsigned sb = ws->sel_bits;
  const int      ne = ws->needed_eq;
  for (int i = blockIdx.x * blockDim.x + threadIdx.x; i < n;
       i += gridDim.x * blockDim.x) {
    const unsigned bits = __float_as_uint(cs[i]);
    if (bits > sb) {
      int p = atomicAdd(&ws->cnt_hi, 1);
      if (p < TOPK) { ws->sel_score[p] = cs[i]; ws->sel_idx[p] = ci[i]; }
    } else if (bits == sb) {
      int p = atomicAdd(&ws->cnt_eq, 1);
      if (p < ne) {
        int slot = TOPK - 1 - p;   // fill from the back; disjoint from cnt_hi
        ws->sel_score[slot] = cs[i]; ws->sel_idx[slot] = ci[i];
      }
    }
  }
}

// ---------------------------------------------------------------------------
// 4) one-block bitonic sort of 2048: score desc, idx asc (top_k tie order)
// ---------------------------------------------------------------------------
__global__ __launch_bounds__(1024)
void k_sort(WS* ws) {
  __shared__ float ss[TOPK];
  __shared__ int   sid[TOPK];
  const unsigned t = threadIdx.x;
  ss[t]         = ws->sel_score[t];
  ss[t + 1024]  = ws->sel_score[t + 1024];
  sid[t]        = ws->sel_idx[t];
  sid[t + 1024] = ws->sel_idx[t + 1024];

  for (unsigned kk = 2; kk <= TOPK; kk <<= 1) {
    for (unsigned j = kk >> 1; j > 0; j >>= 1) {
      __syncthreads();
      const unsigned i = ((t & ~(j - 1u)) << 1) | (t & (j - 1u));
      const unsigned x = i | j;
      const float si = ss[i], sx = ss[x];
      const int   ii = sid[i], ix = sid[x];
      const bool inOrder = (si > sx) || (si == sx && ii <= ix);
      const bool up = ((i & kk) == 0);
      if (inOrder != up) { ss[i] = sx; ss[x] = si; sid[i] = ix; sid[x] = ii; }
    }
  }
  __syncthreads();
  ws->sel_score[t]        = ss[t];
  ws->sel_score[t + 1024] = ss[t + 1024];
  ws->sel_idx[t]          = sid[t];
  ws->sel_idx[t + 1024]   = sid[t + 1024];
}

// ---------------------------------------------------------------------------
// 5) kpts + normalized scores to output
//    layout: kpts[2048][2] | desc[2048][256] | scores[2048]
// ---------------------------------------------------------------------------
__global__ void k_finalize(const WS* __restrict__ ws, float* __restrict__ out) {
  const int t = blockIdx.x * blockDim.x + threadIdx.x;
  if (t >= TOPK) return;
  const int idx = ws->sel_idx[t];
  out[2 * t]     = (float)(idx & (HM_W - 1));
  out[2 * t + 1] = (float)(idx >> 11);
  const float gmax = __int_as_float(ws->gmax_bits);
  out[TOPK * 2 + TOPK * DC + t] = ws->sel_score[t] / gmax;
}

// ---------------------------------------------------------------------------
// 6) bilinear descriptor gather -> f16 A matrix [TOPK][FC]
//    feat fits in the 192MB L2, so this is L2-bandwidth bound.
// ---------------------------------------------------------------------------
__global__ __launch_bounds__(256)
void k_sample(const WS* __restrict__ ws, const float* __restrict__ feat,
              _Float16* __restrict__ rawA) {
  const int kpt = blockIdx.x;
  const int idx = ws->sel_idx[kpt];
  const int xs = idx & (HM_W - 1);
  const int ys = idx >> 11;
  const float fx = ((float)xs + 0.5f) * ((float)FW / (float)HM_W) - 0.5f;
  const float fy = ((float)ys + 0.5f) * ((float)FH / (float)HM_H) - 0.5f;
  const float x0 = floorf(fx), y0 = floorf(fy);
  const float wx = fx - x0,   wy = fy - y0;
  const int x0i = iclamp((int)x0, 0, FW - 1);
  const int x1i = iclamp(x0i + 1, 0, FW - 1);
  const int y0i = iclamp((int)y0, 0, FH - 1);
  const int y1i = iclamp(y0i + 1, 0, FH - 1);
  const float w00 = (1.f - wx) * (1.f - wy);
  const float w10 = wx * (1.f - wy);
  const float w01 = (1.f - wx) * wy;
  const float w11 = wx * wy;

  for (int c = threadIdx.x; c < FC; c += blockDim.x) {
    const float* f = feat + c * (FH * FW);
    const float v = f[y0i * FW + x0i] * w00 + f[y0i * FW + x1i] * w10 +
                    f[y1i * FW + x0i] * w01 + f[y1i * FW + x1i] * w11;
    rawA[kpt * FC + c] = (_Float16)v;
  }
}

// ---------------------------------------------------------------------------
// 7) Wd f32 [DC][FC] -> f16 (already B-fragment friendly: n-major rows of k)
// ---------------------------------------------------------------------------
__global__ void k_cvtW(const float* __restrict__ Wd, _Float16* __restrict__ WdH) {
  const int i = blockIdx.x * blockDim.x + threadIdx.x;
  if (i < DC * FC) WdH[i] = (_Float16)Wd[i];
}

// ---------------------------------------------------------------------------
// 8) GEMM desc = rawA[2048,512] x Wd^T[512,256] + bd, then row L2-normalize.
//    128 blocks x 16 waves; wave (mtile, ntile=wave) owns one 16x16 tile.
//    K advanced 32/step via v_wmma_f32_16x16x32_f16; bias folded into C init.
//    A fragment (contiguous k order chosen here; B uses the same): lane l, row = l%16,
//      elems e -> k = (l/16)*16 + e
//    B fragment: lane l, col n = l%16, elems e -> k = (l/16)*16 + e
//      (Wd row-major [n][k] gives 16 contiguous halves -> two b128 loads)
// ---------------------------------------------------------------------------
__global__ __launch_bounds__(512)
void k_gemm(const _Float16* __restrict__ A, const _Float16* __restrict__ B,
            const float* __restrict__ bd, float* __restrict__ descOut) {
  const int lane = threadIdx.x & 31;
  const int wave = threadIdx.x >> 5;       // 0..15 -> n tile
  const int half = lane >> 4;
  const int l16  = lane & 15;
  const int mbase = blockIdx.x * 16;
  const int nbase = wave * 16;

  const _Float16* arow = A + (size_t)(mbase + l16) * FC + half * 16;
  const _Float16* brow = B + (size_t)(nbase + l16) * FC + half * 16;

  v8f acc;
  const float bias = bd[nbase + l16];
#pragma unroll
  for (int v = 0; v < 8; ++v) acc[v] = bias;

#pragma unroll
  for (int kb = 0; kb < FC; kb += 32) {
    const v8h a0 = *(const v8h*)(arow + kb);
    const v8h a1 = *(const v8h*)(arow + kb + 8);
    const v8h b0 = *(const v8h*)(brow + kb);
    const v8h b1 = *(const v8h*)(brow + kb + 8);
    const v16h af = __builtin_shufflevector(a0, a1, 0, 1, 2, 3, 4, 5, 6, 7,
                                            8, 9, 10, 11, 12, 13, 14, 15);
    const v16h bf = __builtin_shufflevector(b0, b1, 0, 1, 2, 3, 4, 5, 6, 7,
                                            8, 9, 10, 11, 12, 13, 14, 15);
    acc = __builtin_amdgcn_wmma_f32_16x16x32_f16(
        /*neg_a=*/false, af, /*neg_b=*/false, bf,
        /*c_mod=*/(short)0, acc, /*reuse_a=*/false, /*reuse_b=*/false);
  }

  // deterministic row-norm reduction: wave partials -> LDS -> fixed-order sum
  __shared__ float part[16][16];   // [wave][row m]
  __shared__ float rinv[16];
#pragma unroll
  for (int v = 0; v < 8; ++v) {
    float s = acc[v] * acc[v];
    s += __shfl_xor(s, 1, 32);
    s += __shfl_xor(s, 2, 32);
    s += __shfl_xor(s, 4, 32);
    s += __shfl_xor(s, 8, 32);
    if (l16 == 0) part[wave][v + half * 8] = s;   // lanes 0 and 16
  }
  __syncthreads();
  if (threadIdx.x < 16) {
    float t = 0.f;
#pragma unroll
    for (int w = 0; w < 16; ++w) t += part[w][threadIdx.x];
    rinv[threadIdx.x] = 1.0f / fmaxf(sqrtf(t), 1e-12f);
  }
  __syncthreads();
#pragma unroll
  for (int v = 0; v < 8; ++v) {
    const int m = v + half * 8;
    descOut[(size_t)(mbase + m) * DC + nbase + l16] = acc[v] * rinv[m];
  }
}

// ---------------------------------------------------------------------------
// launch
// ---------------------------------------------------------------------------
extern "C" void kernel_launch(void* const* d_in, const int* in_sizes, int n_in,
                              void* d_out, int out_size, void* d_ws, size_t ws_size,
                              hipStream_t stream) {
  const float* heatmap = (const float*)d_in[0];
  const float* feat    = (const float*)d_in[1];
  const float* Wd      = (const float*)d_in[2];
  const float* bd      = (const float*)d_in[3];
  (void)in_sizes; (void)n_in; (void)out_size; (void)ws_size;

  char* base = (char*)d_ws;
  WS* ws = (WS*)base;
  size_t off = (sizeof(WS) + 255) & ~(size_t)255;
  float*    cand_s = (float*)(base + off);  off += (size_t)MAX_CAND * 4;
  int*      cand_i = (int*)(base + off);    off += (size_t)MAX_CAND * 4;
  _Float16* rawA   = (_Float16*)(base + off); off += (size_t)TOPK * FC * 2;
  _Float16* WdH    = (_Float16*)(base + off);

  float* out = (float*)d_out;

  k_init<<<16, 256, 0, stream>>>(ws);
  k_cvtW<<<(DC * FC + 255) / 256, 256, 0, stream>>>(Wd, WdH);

  k_nms<<<dim3(HM_W / 32, HM_H / 8), dim3(32, 8), 0, stream>>>(heatmap, ws,
                                                               cand_s, cand_i);
  k_hist<<<1024, 256, 0, stream>>>(ws, cand_s, 0);
  k_scan<<<1, 1, 0, stream>>>(ws, 0);
  k_hist<<<1024, 256, 0, stream>>>(ws, cand_s, 1);
  k_scan<<<1, 1, 0, stream>>>(ws, 1);
  k_hist<<<1024, 256, 0, stream>>>(ws, cand_s, 2);
  k_scan<<<1, 1, 0, stream>>>(ws, 2);
  k_compact<<<1024, 256, 0, stream>>>(ws, cand_s, cand_i);
  k_sort<<<1, 1024, 0, stream>>>(ws);
  k_finalize<<<TOPK / 256, 256, 0, stream>>>(ws, out);

  k_sample<<<TOPK, 256, 0, stream>>>(ws, feat, rawA);
  k_gemm<<<TOPK / 16, 512, 0, stream>>>(rawA, WdH, bd, out + TOPK * 2);
}
